// InstillAttention_20684562497992
// MI455X (gfx1250) — matrix operand order repaired and checked by reading, not compiled, not run on hardware
//
#include <hip/hip_runtime.h>
#include <hip/hip_bf16.h>

#define DIMX   1024
#define HEADS  16
#define DHEAD  64
#define INNERD 1024
#define QKVN   3072
#define NTOK   2048
#define BATCH  2
#define ROWS   (BATCH * NTOK)      // 4096
#define EPSLN  1e-5f
#define SCL    0.125f              // 1/sqrt(64)

typedef __attribute__((ext_vector_type(16))) __bf16 bf16x16;
typedef __attribute__((ext_vector_type(8)))  float  f32x8;
typedef __attribute__((ext_vector_type(4)))  unsigned int u32x4;
typedef __attribute__((ext_vector_type(8)))  int i32x8;
typedef __attribute__((ext_vector_type(4)))  int i32x4;
typedef int v4i_vs __attribute__((vector_size(16)));   // matches async-builtin pointee

#define AS1 __attribute__((address_space(1)))
#define AS3 __attribute__((address_space(3)))

#if defined(__gfx1250__) && __has_builtin(__builtin_amdgcn_tensor_load_to_lds) && \
    __has_builtin(__builtin_amdgcn_s_wait_tensorcnt)
#define HAVE_TDM 1
#else
#define HAVE_TDM 0
#endif
#if defined(__gfx1250__) && __has_builtin(__builtin_amdgcn_global_load_async_to_lds_b128) && \
    __has_builtin(__builtin_amdgcn_s_wait_asynccnt)
#define HAVE_ASYNC 1
#else
#define HAVE_ASYNC 0
#endif

union Frag {
    bf16x16 v;
    uint4   q[2];
    __bf16  h[16];
};

__device__ __forceinline__ f32x8 zero8() {
    f32x8 z = {0.f, 0.f, 0.f, 0.f, 0.f, 0.f, 0.f, 0.f};
    return z;
}

__device__ __forceinline__ bf16x16 load_frag(const __bf16* p0, const __bf16* p1) {
    Frag f;
    f.q[0] = *(const uint4*)p0;
    f.q[1] = *(const uint4*)p1;
    return f.v;
}

__device__ __forceinline__ f32x8 wmma_bf16(bf16x16 a, bf16x16 b, f32x8 c) {
    return __builtin_amdgcn_wmma_f32_16x16x32_bf16(false, a, false, b, (short)0, c,
                                                   false, false);
}

#if HAVE_TDM
// 2D TDM load: tile_d1 rows x tile_d0 elems (bf16), row stride 'stride' elems,
// LDS gets padding (pad_amount+1) DWORDs after every (1<<(pad_interval+1)) DWORDs.
__device__ __forceinline__ void tdm_load_2d(unsigned lds_off, const void* gsrc,
                                            unsigned tile_d0, unsigned tile_d1,
                                            unsigned stride,
                                            unsigned pad_interval, unsigned pad_amount) {
    unsigned long long ga = (unsigned long long)(size_t)gsrc;
    u32x4 g0;
    g0[0] = 1u;                                         // count=1, user descriptor
    g0[1] = lds_off;                                    // lds_addr (bytes)
    g0[2] = (unsigned)(ga & 0xFFFFFFFFu);               // global_addr[31:0]
    g0[3] = (unsigned)((ga >> 32) & 0x01FFFFFFu) | (2u << 30);  // addr[56:32], type=2
    i32x8 g1;
    g1[0] = (int)((1u << 16) | (1u << 20) | (pad_interval << 22) | (pad_amount << 25));
    g1[1] = (int)((tile_d0 & 0xFFFFu) << 16);           // tensor_dim0 = tile_d0 (no OOB)
    g1[2] = (int)((tile_d0 >> 16) | ((tile_d1 & 0xFFFFu) << 16));  // tensor_dim1
    g1[3] = (int)((tile_d1 >> 16) | (tile_d0 << 16));   // tile_dim0
    g1[4] = (int)(tile_d1 & 0xFFFFu);                   // tile_dim1, tile_dim2=0
    g1[5] = (int)stride;                                // tensor_dim0_stride[31:0]
    g1[6] = 0;
    g1[7] = 0;
    i32x4 z4 = {0, 0, 0, 0};
#if __clang_major__ >= 23
    i32x8 z8 = {0, 0, 0, 0, 0, 0, 0, 0};
    __builtin_amdgcn_tensor_load_to_lds(g0, g1, z4, z4, z8, 0);
#else
    __builtin_amdgcn_tensor_load_to_lds(g0, g1, z4, z4, 0);
#endif
}
#endif

#if HAVE_ASYNC
// copy 32 bytes global -> LDS via two async b128 ops (ioffset applies to both sides)
__device__ __forceinline__ void async_copy16(const __bf16* g, __bf16* l) {
    __builtin_amdgcn_global_load_async_to_lds_b128((AS1 v4i_vs*)g, (AS3 v4i_vs*)l, 0, 0);
    __builtin_amdgcn_global_load_async_to_lds_b128((AS1 v4i_vs*)g, (AS3 v4i_vs*)l, 16, 0);
}
#endif

// ---------------- LayerNorm (fp32 -> bf16), one wave per 1024-wide row ---------
__global__ void __launch_bounds__(128)
ln_kernel(const float* __restrict__ x, const float* __restrict__ g,
          const float* __restrict__ b, __bf16* __restrict__ xn) {
    int wv = threadIdx.x >> 5, lane = threadIdx.x & 31;
    int row = blockIdx.x * 4 + wv;
    const float4* xr = (const float4*)(x + (size_t)row * DIMX);
    float4 vb[8];
    float s = 0.f, ss = 0.f;
#pragma unroll
    for (int i = 0; i < 8; ++i) {
        float4 v = xr[lane + 32 * i];
        vb[i] = v;
        s  += v.x + v.y + v.z + v.w;
        ss += v.x * v.x + v.y * v.y + v.z * v.z + v.w * v.w;
    }
#pragma unroll
    for (int off = 1; off < 32; off <<= 1) {
        s  += __shfl_xor(s,  off, 32);
        ss += __shfl_xor(ss, off, 32);
    }
    float mean = s * (1.f / DIMX);
    float var  = ss * (1.f / DIMX) - mean * mean;
    float rstd = rsqrtf(var + EPSLN);
    const float4* gr = (const float4*)g;
    const float4* br = (const float4*)b;
    __bf16* orow = xn + (size_t)row * DIMX;
#pragma unroll
    for (int i = 0; i < 8; ++i) {
        int idx = lane + 32 * i;
        float4 v = vb[i], gv = gr[idx], bv = br[idx];
        union { __bf16 h[4]; uint2 u; } o;
        o.h[0] = (__bf16)((v.x - mean) * rstd * gv.x + bv.x);
        o.h[1] = (__bf16)((v.y - mean) * rstd * gv.y + bv.y);
        o.h[2] = (__bf16)((v.z - mean) * rstd * gv.z + bv.z);
        o.h[3] = (__bf16)((v.w - mean) * rstd * gv.w + bv.w);
        *(uint2*)(orow + idx * 4) = o.u;
    }
}

// ---------------- fp32 -> bf16 elementwise (4 per thread) ----------------------
__global__ void cvt_kernel(const float* __restrict__ src, __bf16* __restrict__ dst) {
    int i = blockIdx.x * blockDim.x + threadIdx.x;
    float4 v = ((const float4*)src)[i];
    union { __bf16 h[4]; uint2 u; } o;
    o.h[0] = (__bf16)v.x; o.h[1] = (__bf16)v.y;
    o.h[2] = (__bf16)v.z; o.h[3] = (__bf16)v.w;
    *(uint2*)(dst + (size_t)i * 4) = o.u;
}

// ---------- fp32 [K][N] -> bf16 transposed [N][K], tiled through LDS -----------
__global__ void transpose_kernel(const float* __restrict__ src, __bf16* __restrict__ dst,
                                 int K, int N) {
    __shared__ float tile[32][33];
    int tx = threadIdx.x, ty = threadIdx.y;
    int n0 = blockIdx.x * 32, k0 = blockIdx.y * 32;
#pragma unroll
    for (int i = 0; i < 4; ++i)
        tile[ty + 8 * i][tx] = src[(size_t)(k0 + ty + 8 * i) * N + n0 + tx];
    __syncthreads();
#pragma unroll
    for (int i = 0; i < 4; ++i)
        dst[(size_t)(n0 + ty + 8 * i) * K + k0 + tx] = (__bf16)tile[tx][ty + 8 * i];
}

// ---------------- WMMA GEMM: C[M,N] = A[M,K] @ Bt[N,K]^T -----------------------
// Block = 128 threads (4 waves). Block tile 128(M) x 64(N); wave owns 32 rows.
// LDS B tile quadruple-buffered, staged 3 K-steps ahead by TDM/async; B fragments
// register double-buffered (prefetched one K-step ahead) so the 8-wmma chain
// never waits on fresh LDS loads. A fragments register-pipelined from global.
__global__ void __launch_bounds__(128)
gemm_kernel(const __bf16* __restrict__ A, const __bf16* __restrict__ Bt,
            __bf16* __restrict__ Cb, float* __restrict__ Cf,
            const float* __restrict__ bias, int N, int K) {
#if HAVE_TDM || HAVE_ASYNC
    __shared__ __align__(16) __bf16 sB[4][64 * 40];
#else
    __shared__ __align__(16) __bf16 sB[1][64 * 40];
#endif
    int wv = threadIdx.x >> 5, lane = threadIdx.x & 31;
    int hl = lane >> 4, l16 = lane & 15;
    int n0 = blockIdx.x * 64;
    int m0 = blockIdx.y * 128 + wv * 32;
    int srow = threadIdx.x >> 1, spart = threadIdx.x & 1;

    f32x8 acc[2][4];
#pragma unroll
    for (int rg = 0; rg < 2; ++rg)
#pragma unroll
        for (int j = 0; j < 4; ++j) acc[rg][j] = zero8();

    const __bf16* aptr0 = A + (size_t)(m0 + l16) * K + hl * 8;
    const __bf16* aptr1 = aptr0 + (size_t)16 * K;

    // A fragments for kk = 0
    bf16x16 a0 = load_frag(aptr0, aptr0 + 16);
    bf16x16 a1 = load_frag(aptr1, aptr1 + 16);

#if HAVE_TDM || HAVE_ASYNC
    // ---- prologue: stage buffers 0..2 (kk = 0,32,64); wait until buf0 is in
#if HAVE_TDM
    if (wv == 0) {
        tdm_load_2d((unsigned)(size_t)(AS3 __bf16*)&sB[0][0],
                    Bt + (size_t)n0 * K, 32, 64, (unsigned)K, 3, 3);
        if (K > 32)
            tdm_load_2d((unsigned)(size_t)(AS3 __bf16*)&sB[1][0],
                        Bt + (size_t)n0 * K + 32, 32, 64, (unsigned)K, 3, 3);
        if (K > 64)
            tdm_load_2d((unsigned)(size_t)(AS3 __bf16*)&sB[2][0],
                        Bt + (size_t)n0 * K + 64, 32, 64, (unsigned)K, 3, 3);
        __builtin_amdgcn_s_wait_tensorcnt(2);
    }
    __syncthreads();
#else
    async_copy16(Bt + (size_t)(n0 + srow) * K + spart * 16,
                 &sB[0][srow * 40 + spart * 16]);
    if (K > 32)
        async_copy16(Bt + (size_t)(n0 + srow) * K + 32 + spart * 16,
                     &sB[1][srow * 40 + spart * 16]);
    if (K > 64)
        async_copy16(Bt + (size_t)(n0 + srow) * K + 64 + spart * 16,
                     &sB[2][srow * 40 + spart * 16]);
    __builtin_amdgcn_s_wait_asynccnt(4);
    __syncthreads();
#endif
    // initial B fragments from buffer 0
    bf16x16 bfr[4];
#pragma unroll
    for (int j = 0; j < 4; ++j) {
        const __bf16* bp = &sB[0][(j * 16 + l16) * 40 + hl * 8];
        bfr[j] = load_frag(bp, bp + 16);
    }

    int cur = 0;
    for (int kk = 0; kk < K; kk += 32) {
        bool more  = (kk + 32) < K;    // next compute step exists
        bool more2 = (kk + 64) < K;    // a TDM newer than buf(nxt) may be in flight
        bool more3 = (kk + 96) < K;    // stage target exists
        int nxt = cur + 1; if (nxt >= 4) nxt -= 4;
        int stg = cur + 3; if (stg >= 4) stg -= 4;
        // guarantee buf(nxt) staged: all in-flight except (possibly) buf(cur+2)
#if HAVE_TDM
        if (wv == 0) {
            if (more2) __builtin_amdgcn_s_wait_tensorcnt(1);
            else       __builtin_amdgcn_s_wait_tensorcnt(0);
        }
        __syncthreads();   // everyone past reads of buf(stg) (= prefetch 2 iters ago)
        if (wv == 0 && more3)
            tdm_load_2d((unsigned)(size_t)(AS3 __bf16*)&sB[stg][0],
                        Bt + (size_t)n0 * K + kk + 96, 32, 64, (unsigned)K, 3, 3);
#else
        if (more2) __builtin_amdgcn_s_wait_asynccnt(2);
        else       __builtin_amdgcn_s_wait_asynccnt(0);
        __syncthreads();
        if (more3)
            async_copy16(Bt + (size_t)(n0 + srow) * K + kk + 96 + spart * 16,
                         &sB[stg][srow * 40 + spart * 16]);
#endif
        // register-pipeline next A fragments (global) and next B fragments (LDS)
        bf16x16 a0n = a0, a1n = a1;
        if (more) {
            const __bf16* p0 = aptr0 + kk + 32;
            const __bf16* p1 = aptr1 + kk + 32;
            a0n = load_frag(p0, p0 + 16);
            a1n = load_frag(p1, p1 + 16);
        }
        bf16x16 bn[4];
#pragma unroll
        for (int j = 0; j < 4; ++j) bn[j] = bfr[j];
        if (more) {
            const __bf16* bb = sB[nxt];
#pragma unroll
            for (int j = 0; j < 4; ++j) {
                const __bf16* bp = &bb[(j * 16 + l16) * 40 + hl * 8];
                bn[j] = load_frag(bp, bp + 16);
            }
        }
        // 8 wmma back-to-back on fragments loaded a full iteration ago
#pragma unroll
        for (int j = 0; j < 4; ++j) {
            acc[0][j] = wmma_bf16(a0, bfr[j], acc[0][j]);
            acc[1][j] = wmma_bf16(a1, bfr[j], acc[1][j]);
        }
        a0 = a0n; a1 = a1n;
#pragma unroll
        for (int j = 0; j < 4; ++j) bfr[j] = bn[j];
        cur = nxt;
    }
#else
    // ---- sync fallback: stage+compute with two barriers per K-step
    for (int kk = 0; kk < K; kk += 32) {
        __syncthreads();
        {
            const __bf16* gsrc = Bt + (size_t)(n0 + srow) * K + kk + spart * 16;
            __bf16* ldst = &sB[0][srow * 40 + spart * 16];
            *(uint4*)ldst       = *(const uint4*)gsrc;
            *(uint4*)(ldst + 8) = *(const uint4*)(gsrc + 8);
        }
        __syncthreads();
        bf16x16 a0n = a0, a1n = a1;
        if (kk + 32 < K) {
            const __bf16* p0 = aptr0 + kk + 32;
            const __bf16* p1 = aptr1 + kk + 32;
            a0n = load_frag(p0, p0 + 16);
            a1n = load_frag(p1, p1 + 16);
        }
#pragma unroll
        for (int j = 0; j < 4; ++j) {
            const __bf16* bp = &sB[0][(j * 16 + l16) * 40 + hl * 8];
            bf16x16 b = load_frag(bp, bp + 16);
            acc[0][j] = wmma_bf16(a0, b, acc[0][j]);
            acc[1][j] = wmma_bf16(a1, b, acc[1][j]);
        }
        a0 = a0n; a1 = a1n;
    }
#endif

#pragma unroll
    for (int rg = 0; rg < 2; ++rg)
#pragma unroll
        for (int j = 0; j < 4; ++j)
#pragma unroll
            for (int r = 0; r < 8; ++r) {
                int row = m0 + rg * 16 + r + 8 * hl;
                int col = n0 + j * 16 + l16;
                float v = acc[rg][j][r];
                if (Cf) Cf[(size_t)row * N + col] = v + (bias ? bias[col] : 0.f);
                else    Cb[(size_t)row * N + col] = (__bf16)v;
            }
}

// ---------------- Flash attention: shared softmax over V and AV ----------------
// grid = (NTOK/64, HEADS, BATCH), block = 128 (4 waves x 16 query rows).
// V/AV tile global loads are register-pipelined one key-tile ahead.
__global__ void __launch_bounds__(128)
attn_kernel(const __bf16* __restrict__ qkv, const __bf16* __restrict__ avb,
            __bf16* __restrict__ xctx, __bf16* __restrict__ yctx) {
    __shared__ __align__(16) __bf16 sK[32 * 72];     // K tile row-major, padded
    __shared__ __align__(16) __bf16 sVt[64 * 40];    // V tile transposed [d][key]
    __shared__ __align__(16) __bf16 sAVt[64 * 40];   // AV tile transposed
    __shared__ __align__(16) __bf16 sP[4][16 * 40];  // per-wave P relayout strip

    int wv = threadIdx.x >> 5, lane = threadIdx.x & 31;
    int hl = lane >> 4, l16 = lane & 15;
    int h = blockIdx.y, bb = blockIdx.z;
    int q0 = blockIdx.x * 64 + wv * 16;

    const __bf16* qbase  = qkv + (size_t)bb * NTOK * QKVN + h * DHEAD;
    const __bf16* kbase  = qbase + INNERD;
    const __bf16* vbase  = qbase + 2 * INNERD;
    const __bf16* avbase = avb + (size_t)bb * NTOK * INNERD + h * DHEAD;

    const __bf16* qrow = qbase + (size_t)(q0 + l16) * QKVN + hl * 16;
    bf16x16 qa0 = load_frag(qrow,      qrow + 8);       // d = 0..31
    bf16x16 qa1 = load_frag(qrow + 32, qrow + 40);      // d = 32..63

    f32x8 xacc[4], yacc[4];
#pragma unroll
    for (int dt = 0; dt < 4; ++dt) { xacc[dt] = zero8(); yacc[dt] = zero8(); }
    float mrow[8], lrow[8];
#pragma unroll
    for (int r = 0; r < 8; ++r) { mrow[r] = -1e30f; lrow[r] = 0.f; }

    int skey = threadIdx.x >> 2, spart = threadIdx.x & 3;
    const __bf16* vr0  = vbase  + (size_t)skey * QKVN   + spart * 16;
    const __bf16* ar0  = avbase + (size_t)skey * INNERD + spart * 16;

    // preload V/AV tile 0 into registers
    Frag fv, fa;
    fv.q[0] = *(const uint4*)vr0;       fv.q[1] = *(const uint4*)(vr0 + 8);
    fa.q[0] = *(const uint4*)ar0;       fa.q[1] = *(const uint4*)(ar0 + 8);

    for (int m0 = 0; m0 < NTOK; m0 += 32) {
        __syncthreads();
        {   // stage: async K tile + transpose-store the preloaded V/AV registers
            const __bf16* kr = kbase + (size_t)(m0 + skey) * QKVN + spart * 16;
            __bf16* kd = &sK[skey * 72 + spart * 16];
#if HAVE_ASYNC
            async_copy16(kr, kd);
#else
            *(uint4*)kd       = *(const uint4*)kr;
            *(uint4*)(kd + 8) = *(const uint4*)(kr + 8);
#endif
#pragma unroll
            for (int i = 0; i < 16; ++i) {
                sVt[(spart * 16 + i) * 40 + skey]  = fv.h[i];
                sAVt[(spart * 16 + i) * 40 + skey] = fa.h[i];
            }
#if HAVE_ASYNC
            __builtin_amdgcn_s_wait_asynccnt(0);
#endif
        }
        __syncthreads();

        // pipeline: fetch next tile's V/AV while computing this tile
        if (m0 + 32 < NTOK) {
            const __bf16* vr = vr0 + (size_t)(m0 + 32) * QKVN;
            const __bf16* ar = ar0 + (size_t)(m0 + 32) * INNERD;
            fv.q[0] = *(const uint4*)vr; fv.q[1] = *(const uint4*)(vr + 8);
            fa.q[0] = *(const uint4*)ar; fa.q[1] = *(const uint4*)(ar + 8);
        }

        // scores: preload 4 K fragments, then 4 wmma
        bf16x16 kb[4];
        {
            const __bf16* b0 = &sK[l16 * 72 + hl * 16];
            const __bf16* b1 = &sK[l16 * 72 + 32 + hl * 16];
            const __bf16* b2 = &sK[(16 + l16) * 72 + hl * 16];
            const __bf16* b3 = &sK[(16 + l16) * 72 + 32 + hl * 16];
            kb[0] = load_frag(b0, b0 + 8);
            kb[1] = load_frag(b1, b1 + 8);
            kb[2] = load_frag(b2, b2 + 8);
            kb[3] = load_frag(b3, b3 + 8);
        }
        f32x8 s0 = zero8(), s1 = zero8();
        s0 = wmma_bf16(qa0, kb[0], s0);
        s0 = wmma_bf16(qa1, kb[1], s0);
        s1 = wmma_bf16(qa0, kb[2], s1);
        s1 = wmma_bf16(qa1, kb[3], s1);

        // online softmax: per-row stats in 8 lane-local slots (row = r + 8*hl)
        float p0[8], p1[8], fc[8];
#pragma unroll
        for (int r = 0; r < 8; ++r) {
            float sa = s0[r] * SCL, sb = s1[r] * SCL;
            float mx = fmaxf(sa, sb);
#pragma unroll
            for (int off = 1; off < 16; off <<= 1) mx = fmaxf(mx, __shfl_xor(mx, off, 32));
            float nm = fmaxf(mrow[r], mx);
            float f  = __expf(mrow[r] - nm);
            mrow[r] = nm;
            float pa = __expf(sa - nm), pb = __expf(sb - nm);
            p0[r] = pa; p1[r] = pb;
            float rs = pa + pb;
#pragma unroll
            for (int off = 1; off < 16; off <<= 1) rs += __shfl_xor(rs, off, 32);
            lrow[r] = lrow[r] * f + rs;
            fc[r] = f;
        }
#pragma unroll
        for (int dt = 0; dt < 4; ++dt)
#pragma unroll
            for (int r = 0; r < 8; ++r) { xacc[dt][r] *= fc[r]; yacc[dt][r] *= fc[r]; }

        // relayout P (C-frag -> A-frag) through wave-private LDS strip
        __bf16* sPw = sP[wv];
#pragma unroll
        for (int r = 0; r < 8; ++r) {
            int rr = r + 8 * hl;
            sPw[rr * 40 + l16]      = (__bf16)p0[r];
            sPw[rr * 40 + 16 + l16] = (__bf16)p1[r];
        }
        const __bf16* pp = &sPw[l16 * 40 + hl * 16];
        bf16x16 pa = load_frag(pp, pp + 8);

        // context: preload fragment batches, then uninterrupted wmma chains
        bf16x16 bx[4];
#pragma unroll
        for (int dt = 0; dt < 4; ++dt) {
            const __bf16* p = &sVt[(dt * 16 + l16) * 40 + hl * 16];
            bx[dt] = load_frag(p, p + 8);
        }
#pragma unroll
        for (int dt = 0; dt < 4; ++dt) xacc[dt] = wmma_bf16(pa, bx[dt], xacc[dt]);
        bf16x16 by[4];
#pragma unroll
        for (int dt = 0; dt < 4; ++dt) {
            const __bf16* p = &sAVt[(dt * 16 + l16) * 40 + hl * 16];
            by[dt] = load_frag(p, p + 8);
        }
#pragma unroll
        for (int dt = 0; dt < 4; ++dt) yacc[dt] = wmma_bf16(pa, by[dt], yacc[dt]);
    }

    // finalize: divide by softmax denominator, write merged-head bf16 contexts
#pragma unroll
    for (int r = 0; r < 8; ++r) {
        float inv = 1.0f / lrow[r];
        int row = q0 + r + 8 * hl;
        size_t base = ((size_t)bb * NTOK + row) * INNERD + h * DHEAD;
#pragma unroll
        for (int dt = 0; dt < 4; ++dt) {
            int col = dt * 16 + l16;
            xctx[base + col] = (__bf16)(xacc[dt][r] * inv);
            yctx[base + col] = (__bf16)(yacc[dt][r] * inv);
        }
    }
}

extern "C" void kernel_launch(void* const* d_in, const int* in_sizes, int n_in,
                              void* d_out, int out_size, void* d_ws, size_t ws_size,
                              hipStream_t stream) {
    const float* x      = (const float*)d_in[0];
    const float* y      = (const float*)d_in[1];
    const float* ln_g   = (const float*)d_in[2];
    const float* ln_b   = (const float*)d_in[3];
    const float* w_qkv  = (const float*)d_in[4];
    const float* w_av   = (const float*)d_in[5];
    const float* w_out  = (const float*)d_in[6];
    const float* b_out  = (const float*)d_in[7];
    const float* w_yout = (const float*)d_in[8];
    const float* b_yout = (const float*)d_in[9];
    float* out = (float*)d_out;

    char* ws = (char*)d_ws;
    size_t off = 0;
    auto alloc = [&](size_t elems) { __bf16* p = (__bf16*)(ws + off); off += elems * 2; return p; };
    __bf16* xn     = alloc((size_t)ROWS * DIMX);
    __bf16* ybf    = alloc((size_t)ROWS * DIMX);
    __bf16* wqkvT  = alloc((size_t)QKVN * DIMX);
    __bf16* wavT   = alloc((size_t)INNERD * DIMX);
    __bf16* woutT  = alloc((size_t)DIMX * INNERD);
    __bf16* wyoutT = alloc((size_t)DIMX * INNERD);
    __bf16* qkvb   = alloc((size_t)ROWS * QKVN);
    __bf16* avbuf  = alloc((size_t)ROWS * INNERD);
    __bf16* xctx   = alloc((size_t)ROWS * INNERD);
    __bf16* yctx   = alloc((size_t)ROWS * INNERD);

    dim3 tb(32, 8);
    ln_kernel<<<ROWS / 4, 128, 0, stream>>>(x, ln_g, ln_b, xn);
    cvt_kernel<<<((size_t)ROWS * DIMX / 4) / 256, 256, 0, stream>>>(y, ybf);
    transpose_kernel<<<dim3(QKVN / 32, DIMX / 32), tb, 0, stream>>>(w_qkv, wqkvT, DIMX, QKVN);
    transpose_kernel<<<dim3(INNERD / 32, DIMX / 32), tb, 0, stream>>>(w_av, wavT, DIMX, INNERD);
    transpose_kernel<<<dim3(DIMX / 32, INNERD / 32), tb, 0, stream>>>(w_out, woutT, INNERD, DIMX);
    transpose_kernel<<<dim3(DIMX / 32, INNERD / 32), tb, 0, stream>>>(w_yout, wyoutT, INNERD, DIMX);

    gemm_kernel<<<dim3(QKVN / 64, ROWS / 128), 128, 0, stream>>>(
        xn, wqkvT, qkvb, nullptr, nullptr, QKVN, DIMX);
    gemm_kernel<<<dim3(INNERD / 64, ROWS / 128), 128, 0, stream>>>(
        ybf, wavT, avbuf, nullptr, nullptr, INNERD, DIMX);

    attn_kernel<<<dim3(NTOK / 64, HEADS, BATCH), 128, 0, stream>>>(qkvb, avbuf, xctx, yctx);

    gemm_kernel<<<dim3(DIMX / 64, ROWS / 128), 128, 0, stream>>>(
        xctx, woutT, nullptr, out, b_out, DIMX, INNERD);
    gemm_kernel<<<dim3(DIMX / 64, ROWS / 128), 128, 0, stream>>>(
        yctx, wyoutT, nullptr, out + (size_t)ROWS * DIMX, b_yout, DIMX, INNERD);
}
